// CrossFrameAttention_7009386627548
// MI455X (gfx1250) — matrix operand order repaired and checked by reading, not compiled, not run on hardware
//
#include <hip/hip_runtime.h>
#include <hip/hip_bf16.h>

// ---------------- types for WMMA ----------------
typedef __attribute__((ext_vector_type(16))) __bf16 v16bf;
typedef __attribute__((ext_vector_type(8)))  float  v8f;

union FragB {
  v16bf v;
  uint4 q[2];
};

__device__ __forceinline__ unsigned short f2bf(float f) {
  unsigned u = __float_as_uint(f);
  u += 0x7FFFu + ((u >> 16) & 1u);        // round-to-nearest-even
  return (unsigned short)(u >> 16);
}
__device__ __forceinline__ float bf2f(unsigned short h) {
  return __uint_as_float(((unsigned)h) << 16);
}

// ---------------- geometry ----------------
#define C_DIM   128
#define N_FR    8          // frames (sequence length)
#define NHEADS  8
#define HD      16
#define TP      16         // spatial positions per workgroup (tile along W)
#define ROWS    128        // TP * N_FR rows of the token-block GEMMs
#define PITCH_A 132        // fp32 residual/raw buffer pitch (floats)
#define PITCH_X 136        // bf16 x_ln / attn_out pitch (ushorts)
#define HS      132        // qkv: elems per (p,h) head-block (128 + pad)
#define PS      (8*HS)     // qkv: elems per position p          (1056)
#define QS      (16*PS)    // qkv: elems per q/k/v selector      (16896)

#define SMEM_A  (ROWS*PITCH_A*4)          // 67584
#define SMEM_X  (ROWS*PITCH_X*2)          // 34816
#define SMEM_Q  (3*QS*2)                  // 101376
#define SMEM_TOTAL (SMEM_A + SMEM_X + SMEM_Q)  // 203776 bytes

// ---------------- prologue: weights -> bf16, transposed [N][K] ----------------
__global__ void prep_weights(const float* __restrict__ w_qkv,   // [128][384]
                             const float* __restrict__ w_proj,  // [128][128]
                             unsigned short* __restrict__ wqkvT, // [384][128] bf16
                             unsigned short* __restrict__ wprojT)// [128][128] bf16
{
  int r = blockIdx.x;         // 0..511
  int t = threadIdx.x;        // 0..127  (= K index)
  if (r < 384) {
    wqkvT[r * 128 + t] = f2bf(w_qkv[t * 384 + r]);
  } else {
    int rp = r - 384;
    wprojT[rp * 128 + t] = f2bf(w_proj[t * 128 + rp]);
  }
}

// ---------------- main fused kernel ----------------
__global__ __launch_bounds__(256, 1)
void cfa_main(const float* __restrict__ feat,
              const float* __restrict__ gamma,
              const float* __restrict__ beta,
              const float* __restrict__ bproj,
              const unsigned short* __restrict__ wqkvT,
              const unsigned short* __restrict__ wprojT,
              float* __restrict__ outp)
{
  extern __shared__ __align__(16) char smem_raw[];
  float*          bufA = (float*)smem_raw;                       // raw fp32 / residual / final out
  unsigned short* xln  = (unsigned short*)(smem_raw + SMEM_A);   // bf16 LN-out, later attn-out
  unsigned short* qkv  = (unsigned short*)(smem_raw + SMEM_A + SMEM_X);

  const int tid  = threadIdx.x;
  const int wave = tid >> 5;       // 8 waves (wave32)
  const int lane = tid & 31;
  const int half = (lane >> 4) & 1;
  const int lrow = lane & 15;

  const int wt   = blockIdx.x;     // 0..5   (W tile)
  const int hrow = blockIdx.y;     // 0..95
  const int b    = blockIdx.z;     // 0..3
  const int w0   = wt * 16;

  // element (n,c,p): baseB + (n*128+c)*9216 + p
  const long long baseB = (long long)b * (8LL * 128LL * 9216LL)
                        + (long long)hrow * 96LL + (long long)w0;

  // ---- phase 1: coalesced load of raw features -> LDS (also the residual) ----
  for (int it = 0; it < 16; ++it) {
    int qi    = it * 256 + tid;            // 0..4095 quarter-rows
    int rowid = qi >> 2;                   // (n,c)
    int part  = qi & 3;
    int n     = rowid >> 7;
    int c     = rowid & 127;
    const float4 vv = *(const float4*)(feat + baseB + (long long)(n * 128 + c) * 9216LL + part * 4);
    int p0 = part * 4;
    bufA[(n * 16 + p0 + 0) * PITCH_A + c] = vv.x;
    bufA[(n * 16 + p0 + 1) * PITCH_A + c] = vv.y;
    bufA[(n * 16 + p0 + 2) * PITCH_A + c] = vv.z;
    bufA[(n * 16 + p0 + 3) * PITCH_A + c] = vv.w;
  }
  __syncthreads();

  // ---- phase 2: LayerNorm over C=128, emit bf16 ----
  {
    int row = tid >> 1;                    // 0..127  (row = n*16 + p)
    int hf  = tid & 1;
    const float* rp = bufA + row * PITCH_A + hf * 64;
    float s = 0.f, s2 = 0.f;
    for (int i = 0; i < 64; ++i) { float x = rp[i]; s += x; s2 += x * x; }
    s  += __shfl_xor(s, 1, 32);
    s2 += __shfl_xor(s2, 1, 32);
    float mu  = s * (1.0f / 128.0f);
    float var = s2 * (1.0f / 128.0f) - mu * mu;
    float rs  = __frsqrt_rn(var + 1e-5f);
    unsigned short* xp = xln + row * PITCH_X + hf * 64;
    for (int i = 0; i < 64; ++i) {
      int c = hf * 64 + i;
      xp[i] = f2bf((rp[i] - mu) * rs * gamma[c] + beta[c]);
    }
  }
  __syncthreads();

  // ---- phase 3: qkv GEMM  (128x128) @ (128x384) via v_wmma_f32_16x16x32_bf16 ----
  {
    const int m0 = wave * 16;              // wave <-> frame n
    FragB a[4];
#pragma unroll
    for (int kk = 0; kk < 4; ++kk) {       // ISA A-layout: lanes 0-15 K{0-7,16-23}; 16-31 K{8-15,24-31}
      const unsigned short* base = xln + (m0 + lrow) * PITCH_X + kk * 32 + half * 8;
      a[kk].q[0] = *(const uint4*)(base);
      a[kk].q[1] = *(const uint4*)(base + 16);
    }
    for (int nt = 0; nt < 24; ++nt) {
      int col = nt * 16 + lrow;
      v8f acc = {};
#pragma unroll
      for (int kk = 0; kk < 4; ++kk) {     // B-layout: same as A, lanes 0-15 K{0-7,16-23}; 16-31 K{8-15,24-31}
        FragB bf;
        const unsigned short* wb = wqkvT + col * 128 + kk * 32 + half * 8;
        bf.q[0] = *(const uint4*)(wb);
        bf.q[1] = *(const uint4*)(wb + 16);
        acc = __builtin_amdgcn_wmma_f32_16x16x32_bf16(false, a[kk].v, false, bf.v,
                                                      (short)0, acc, false, false);
      }
      // scatter D tile into qkv LDS: [sel][p][h][n][d], n == wave
      int sel = col >> 7;
      int rem = col & 127;
      int hh  = rem >> 4;
      int dd  = rem & 15;
      unsigned short* qb = qkv + sel * QS + hh * HS + wave * 16 + dd;
#pragma unroll
      for (int j = 0; j < 8; ++j) {
        int p = j + half * 8;              // D-layout: VGPR j -> M = j + 8*half
        qb[p * PS] = f2bf(acc[j]);
      }
    }
  }
  __syncthreads();

  // ---- phase 4: 8x8 attention + softmax per (position, head), VALU fp32 ----
  if (tid < 128) {
    int p = tid >> 3;
    int h = tid & 7;
    const unsigned short* qb = qkv + 0 * QS + p * PS + h * HS;
    const unsigned short* kb = qkv + 1 * QS + p * PS + h * HS;
    const unsigned short* vb = qkv + 2 * QS + p * PS + h * HS;
    for (int n = 0; n < 8; ++n) {
      float qr[16];
#pragma unroll
      for (int d = 0; d < 16; ++d) qr[d] = bf2f(qb[n * 16 + d]);
      float s[8];
      float mx = -1e30f;
      for (int m = 0; m < 8; ++m) {
        float acc = 0.f;
#pragma unroll
        for (int d = 0; d < 16; ++d) acc += qr[d] * bf2f(kb[m * 16 + d]);
        acc *= 0.25f;                      // HEAD_DIM^-0.5
        s[m] = acc;
        mx = fmaxf(mx, acc);
      }
      float sum = 0.f;
      for (int m = 0; m < 8; ++m) { s[m] = __expf(s[m] - mx); sum += s[m]; }
      float inv = 1.0f / sum;
      float o[16];
#pragma unroll
      for (int d = 0; d < 16; ++d) o[d] = 0.f;
      for (int m = 0; m < 8; ++m) {
        float pm = s[m] * inv;
#pragma unroll
        for (int d = 0; d < 16; ++d) o[d] += pm * bf2f(vb[m * 16 + d]);
      }
      unsigned short* ob = xln + (n * 16 + p) * PITCH_X + h * 16;  // reuse xln as attn-out
#pragma unroll
      for (int d = 0; d < 16; ++d) ob[d] = f2bf(o[d]);
    }
  }
  __syncthreads();

  // ---- phase 5: proj GEMM (128x128)@(128x128) + b_proj + residual (in bufA) ----
  {
    const int m0 = wave * 16;
    FragB a[4];
#pragma unroll
    for (int kk = 0; kk < 4; ++kk) {
      const unsigned short* base = xln + (m0 + lrow) * PITCH_X + kk * 32 + half * 8;
      a[kk].q[0] = *(const uint4*)(base);
      a[kk].q[1] = *(const uint4*)(base + 16);
    }
    for (int nt = 0; nt < 8; ++nt) {
      int col = nt * 16 + lrow;
      v8f acc = {};
#pragma unroll
      for (int kk = 0; kk < 4; ++kk) {
        FragB bf;
        const unsigned short* wb = wprojT + col * 128 + kk * 32 + half * 8;
        bf.q[0] = *(const uint4*)(wb);
        bf.q[1] = *(const uint4*)(wb + 16);
        acc = __builtin_amdgcn_wmma_f32_16x16x32_bf16(false, a[kk].v, false, bf.v,
                                                      (short)0, acc, false, false);
      }
      float bb = bproj[col];
#pragma unroll
      for (int j = 0; j < 8; ++j) {
        int row = m0 + j + half * 8;       // only this wave's rows -> no race
        float* ap = bufA + row * PITCH_A + col;
        *ap = *ap + bb + acc[j];           // residual + proj + bias, fp32
      }
    }
  }
  __syncthreads();

  // ---- phase 6: coalesced store back to (B,N,C,H,W) ----
  for (int it = 0; it < 16; ++it) {
    int qi    = it * 256 + tid;
    int rowid = qi >> 2;
    int part  = qi & 3;
    int n     = rowid >> 7;
    int c     = rowid & 127;
    int p0    = part * 4;
    float4 vv;
    vv.x = bufA[(n * 16 + p0 + 0) * PITCH_A + c];
    vv.y = bufA[(n * 16 + p0 + 1) * PITCH_A + c];
    vv.z = bufA[(n * 16 + p0 + 2) * PITCH_A + c];
    vv.w = bufA[(n * 16 + p0 + 3) * PITCH_A + c];
    *(float4*)(outp + baseB + (long long)(n * 128 + c) * 9216LL + p0) = vv;
  }
}

extern "C" void kernel_launch(void* const* d_in, const int* in_sizes, int n_in,
                              void* d_out, int out_size, void* d_ws, size_t ws_size,
                              hipStream_t stream) {
  (void)in_sizes; (void)n_in; (void)out_size; (void)ws_size;
  const float* feat   = (const float*)d_in[0];
  const float* w_qkv  = (const float*)d_in[1];
  const float* w_proj = (const float*)d_in[2];
  const float* bproj  = (const float*)d_in[3];
  const float* gamma  = (const float*)d_in[4];
  const float* beta   = (const float*)d_in[5];
  float* outp = (float*)d_out;

  unsigned short* wqkvT  = (unsigned short*)d_ws;                                 // 384*128 bf16
  unsigned short* wprojT = (unsigned short*)((char*)d_ws + 384 * 128 * sizeof(unsigned short));

  hipLaunchKernelGGL(prep_weights, dim3(512), dim3(128), 0, stream,
                     w_qkv, w_proj, wqkvT, wprojT);

  hipLaunchKernelGGL(cfa_main, dim3(6, 96, 4), dim3(256), SMEM_TOTAL, stream,
                     feat, gamma, beta, bproj, wqkvT, wprojT, outp);
}
